// YuanMoeLayer_9483287790023
// MI455X (gfx1250) — compile-verified, matrix operand order checked by reading, not run on hardware
//
#include <hip/hip_runtime.h>
#include <hip/hip_bf16.h>

// Problem constants (from reference): T=2048, H=2048, E=16, I=1408, K=2
#define T_TOK 2048
#define H_DIM 2048
#define E_EXP 16
#define I_DIM 1408

typedef __attribute__((ext_vector_type(16))) __bf16 v16bf;
typedef __attribute__((ext_vector_type(8)))  float  v8f;

union FragBF { v16bf v; unsigned int u[8]; };

// Async global->LDS staging (gfx1250 GLOBAL_LOAD_ASYNC_TO_LDS_B128, ASYNCcnt)
#if defined(__has_builtin)
#  if __has_builtin(__builtin_amdgcn_global_load_async_to_lds_b128)
#    define HAVE_ASYNC_LDS 1
#  endif
#endif
#ifndef HAVE_ASYNC_LDS
#  define HAVE_ASYNC_LDS 0
#endif

// builtin parameter type per hipcc diagnostic: (vector_size(16) int)*
typedef int v4i_ls __attribute__((vector_size(16)));

// ---------- helpers ----------
__device__ __forceinline__ unsigned short f2bf(float f) {
    unsigned int u = __float_as_uint(f);
    u += 0x7fffu + ((u >> 16) & 1u);      // round-to-nearest-even
    return (unsigned short)(u >> 16);
}
__device__ __forceinline__ unsigned int pack_bf2(float lo, float hi) {
    return (unsigned int)f2bf(lo) | ((unsigned int)f2bf(hi) << 16);
}
// 16-bit A/B fragment K-mapping (ISA 16-bit A 16x32 layout):
// lanes 0-15: K {0..7} in v0-3, {16..23} in v4-7 ; lanes 16-31: K {8..15},{24..31}
__device__ __forceinline__ int kmap(int vr, int hf) {
    int j = 2 * vr;
    return j + (j < 8 ? (hf ? 8 : 0) : (hf ? 16 : 8));
}
// stage 32 bytes (16 bf16) global -> LDS (async DMA when available)
__device__ __forceinline__ void stage16h(const unsigned short* src, unsigned short* dst) {
#if HAVE_ASYNC_LDS
    __builtin_amdgcn_global_load_async_to_lds_b128((v4i_ls*)src, (v4i_ls*)dst, 0, 0);
    __builtin_amdgcn_global_load_async_to_lds_b128((v4i_ls*)(src + 8), (v4i_ls*)(dst + 8), 0, 0);
#else
    uint4* d = (uint4*)dst;
    d[0] = *(const uint4*)(src);
    d[1] = *(const uint4*)(src + 8);
#endif
}
__device__ __forceinline__ void async_fence() {
#if HAVE_ASYNC_LDS
    asm volatile("s_wait_asynccnt 0x0" ::: "memory");
#endif
}
// load 16 fp32 into registers
__device__ __forceinline__ void loadB16f(const float* src, float4& a, float4& b,
                                         float4& c, float4& d) {
    const float4* p = (const float4*)src;
    a = p[0]; b = p[1]; c = p[2]; d = p[3];
}
// convert+pack 16 fp32 -> 16 bf16 and store to LDS
__device__ __forceinline__ void storeB16(unsigned short* dst, float4 a, float4 b,
                                         float4 c, float4 d) {
    uint4* p = (uint4*)dst;
    p[0] = make_uint4(pack_bf2(a.x, a.y), pack_bf2(a.z, a.w),
                      pack_bf2(b.x, b.y), pack_bf2(b.z, b.w));
    p[1] = make_uint4(pack_bf2(c.x, c.y), pack_bf2(c.z, c.w),
                      pack_bf2(d.x, d.y), pack_bf2(d.z, d.w));
}

// ---------- small kernels ----------
__global__ void init_kernel(float* y, int* cnt, int n) {
    int i = blockIdx.x * 256 + threadIdx.x;
    if (i < n) y[i] = 0.0f;
    if (blockIdx.x == 0 && threadIdx.x < E_EXP) cnt[threadIdx.x] = 0;
}

__global__ void cvt_kernel(const float* __restrict__ x, unsigned short* __restrict__ xb, int n) {
    int i = blockIdx.x * 256 + threadIdx.x;
    if (i < n) xb[i] = f2bf(x[i]);
}

// mix[t, c] = sum_h x[t,h] * wqkv[h, c]   (c in 0..47)
__global__ void mix_kernel(const float* __restrict__ x, const float* __restrict__ wqkv,
                           float* __restrict__ mix) {
    int idx = blockIdx.x * 256 + threadIdx.x;
    if (idx >= T_TOK * 48) return;
    int t = idx / 48, c = idx % 48;
    const float* xr = x + (size_t)t * H_DIM;
    float acc = 0.0f;
    for (int h = 0; h < H_DIM; ++h) acc = fmaf(xr[h], wqkv[h * 48 + c], acc);
    mix[idx] = acc;
}

// per-token: attention router -> softmax -> top-2 -> renormalize
__global__ void router_kernel(const float* __restrict__ mix, float* __restrict__ combine,
                              int* __restrict__ ids, int* __restrict__ cnt) {
    int t = blockIdx.x * 256 + threadIdx.x;
    if (t >= T_TOK) return;
    const float* m = mix + (size_t)t * 48;
    float q[16], kk[16], vv[16], lg[16];
#pragma unroll
    for (int e = 0; e < 16; ++e) { q[e] = m[e]; kk[e] = m[16 + e]; vv[e] = m[32 + e]; }
#pragma unroll
    for (int e = 0; e < 16; ++e) {
        float mx = -3.0e38f;
#pragma unroll
        for (int f = 0; f < 16; ++f) mx = fmaxf(mx, q[e] * kk[f]);
        float s = 0.0f, sv = 0.0f;
#pragma unroll
        for (int f = 0; f < 16; ++f) {
            float p = __expf(q[e] * kk[f] - mx);
            s += p; sv = fmaf(p, vv[f], sv);
        }
        lg[e] = sv * __builtin_amdgcn_rcpf(s);
    }
    float mm = -3.0e38f;
#pragma unroll
    for (int e = 0; e < 16; ++e) mm = fmaxf(mm, lg[e]);
    float p[16];
#pragma unroll
    for (int e = 0; e < 16; ++e) p[e] = __expf(lg[e] - mm);
    // top-2 (first index wins ties, like lax.top_k); renorm ratio is norm-free
    int i1 = 0;
#pragma unroll
    for (int e = 1; e < 16; ++e) if (p[e] > p[i1]) i1 = e;
    int i2 = (i1 == 0) ? 1 : 0;
#pragma unroll
    for (int e = 0; e < 16; ++e) if (e != i1 && p[e] > p[i2]) i2 = e;
    float rws = __builtin_amdgcn_rcpf(p[i1] + p[i2]);
#pragma unroll
    for (int e = 0; e < 16; ++e) combine[t * E_EXP + e] = 0.0f;
    combine[t * E_EXP + i1] = p[i1] * rws;
    combine[t * E_EXP + i2] = p[i2] * rws;
    ids[2 * t] = i1; ids[2 * t + 1] = i2;
    atomicAdd(&cnt[i1], 1); atomicAdd(&cnt[i2], 1);
}

__global__ void scan_kernel(const int* __restrict__ cnt, int* __restrict__ segoff,
                            int* __restrict__ cur) {
    if (blockIdx.x == 0 && threadIdx.x == 0) {
        int run = 0;
        for (int e = 0; e < E_EXP; ++e) { segoff[e] = run; cur[e] = run; run += cnt[e]; }
    }
}

__global__ void fill_kernel(const int* __restrict__ ids, int* __restrict__ cur,
                            int* __restrict__ perm) {
    int t = blockIdx.x * 256 + threadIdx.x;
    if (t >= T_TOK) return;
    int e1 = ids[2 * t], e2 = ids[2 * t + 1];
    perm[atomicAdd(&cur[e1], 1)] = t;
    perm[atomicAdd(&cur[e2], 1)] = t;
}

// ---------- GEMM1: g = x @ w1e^T, fused SwiGLU -> hb (bf16) ----------
// grid: (I/64, T/64, E), block 128 (4 waves). Tile 64 tokens x 64 gate cols
// (+ the matching 64 up cols), K over H in steps of 32, double-buffered LDS:
// async A-DMA and B global loads for tile k+1 overlap the WMMAs on tile k.
__global__ __launch_bounds__(128)
void gemm1_kernel(const unsigned short* __restrict__ xb, const float* __restrict__ w1,
                  const int* __restrict__ perm, const int* __restrict__ segoff,
                  const int* __restrict__ cnt, unsigned short* __restrict__ hb) {
    __shared__ unsigned short As[2][64 * 32];
    __shared__ unsigned short Bg[2][64 * 32];
    __shared__ unsigned short Bu[2][64 * 32];
    __shared__ int toks[64];

    const int e  = blockIdx.z;
    const int Ne = cnt[e];
    const int m0 = blockIdx.y * 64;
    if (m0 >= Ne) return;
    const int n0 = blockIdx.x * 64;
    const int so = segoff[e];
    const int tid = threadIdx.x;
    const int lane = tid & 31, wid = tid >> 5;
    const int lr = lane & 15, hf = lane >> 4;
    const int srow = tid >> 1, sseg = (tid & 1) * 16;
    const int soff = srow * 32 + sseg;

    if (tid < 64) {
        int gm = m0 + tid;
        toks[tid] = (gm < Ne) ? perm[so + gm] : perm[so];
    }
    __syncthreads();

    const unsigned short* arow_src = xb + (size_t)toks[srow] * H_DIM + sseg;
    const float* grow_src = w1 + ((size_t)e * 2 * I_DIM + (n0 + srow)) * H_DIM + sseg;
    const float* urow_src = grow_src + (size_t)I_DIM * H_DIM;

    v8f zero = {};
    v8f accg[4], accu[4];
#pragma unroll
    for (int i = 0; i < 4; ++i) { accg[i] = zero; accu[i] = zero; }

    // prologue: stage tile 0 into buffer 0
    stage16h(arow_src, &As[0][soff]);
    {
        float4 a, b, c, d;
        loadB16f(grow_src, a, b, c, d); storeB16(&Bg[0][soff], a, b, c, d);
        loadB16f(urow_src, a, b, c, d); storeB16(&Bu[0][soff], a, b, c, d);
    }
    async_fence();
    __syncthreads();

    int buf = 0;
    for (int k0 = 0; k0 < H_DIM; k0 += 32, buf ^= 1) {
        const int kn = k0 + 32;
        const bool hasnext = kn < H_DIM;
        float4 g0, g1, g2, g3, u0, u1, u2, u3;
        if (hasnext) {
            // issue next tile: A via async DMA, B globals into registers
            stage16h(arow_src + kn, &As[buf ^ 1][soff]);
            loadB16f(grow_src + kn, g0, g1, g2, g3);
            loadB16f(urow_src + kn, u0, u1, u2, u3);
            if (kn + 32 < H_DIM) {           // prefetch 2 tiles ahead (WGP scope)
                __builtin_prefetch(grow_src + kn + 32, 0, 3);
                __builtin_prefetch(urow_src + kn + 32, 0, 3);
            }
        }

        // compute on current buffer
        FragBF af;
        const int arow = wid * 16 + lr;
#pragma unroll
        for (int vr = 0; vr < 8; ++vr)
            af.u[vr] = *(const unsigned int*)&As[buf][arow * 32 + kmap(vr, hf)];
#pragma unroll
        for (int nb = 0; nb < 4; ++nb) {
            FragBF bg, bu;
            const int brow = nb * 16 + lr;
#pragma unroll
            for (int vr = 0; vr < 8; ++vr) {
                int kx = kmap(vr, hf);
                bg.u[vr] = *(const unsigned int*)&Bg[buf][brow * 32 + kx];
                bu.u[vr] = *(const unsigned int*)&Bu[buf][brow * 32 + kx];
            }
            accg[nb] = __builtin_amdgcn_wmma_f32_16x16x32_bf16(
                false, af.v, false, bg.v, (short)0, accg[nb], false, false);
            accu[nb] = __builtin_amdgcn_wmma_f32_16x16x32_bf16(
                false, af.v, false, bu.v, (short)0, accu[nb], false, false);
        }

        if (hasnext) {
            storeB16(&Bg[buf ^ 1][soff], g0, g1, g2, g3);
            storeB16(&Bu[buf ^ 1][soff], u0, u1, u2, u3);
        }
        async_fence();
        __syncthreads();
    }

    // fused SwiGLU epilogue: h = silu(gate) * up -> bf16
#pragma unroll
    for (int nb = 0; nb < 4; ++nb) {
#pragma unroll
        for (int r = 0; r < 8; ++r) {
            int row = wid * 16 + r + 8 * hf;
            int gm = m0 + row;
            if (gm < Ne) {
                float g = accg[nb][r], u = accu[nb][r];
                float s = g * __builtin_amdgcn_rcpf(1.0f + __expf(-g));
                hb[(size_t)(so + gm) * I_DIM + (n0 + nb * 16 + lr)] = f2bf(s * u);
            }
        }
    }
}

// ---------- GEMM2: y[t,:] += ce * (h @ w2e^T) ----------
// grid: (H/64, T/64, E), block 128, same double-buffered pipeline.
__global__ __launch_bounds__(128)
void gemm2_kernel(const unsigned short* __restrict__ hb, const float* __restrict__ w2,
                  const int* __restrict__ perm, const int* __restrict__ segoff,
                  const int* __restrict__ cnt, const float* __restrict__ combine,
                  float* __restrict__ y) {
    __shared__ unsigned short As[2][64 * 32];
    __shared__ unsigned short Bs[2][64 * 32];
    __shared__ int toks[64];
    __shared__ float ceL[64];

    const int e  = blockIdx.z;
    const int Ne = cnt[e];
    const int m0 = blockIdx.y * 64;
    if (m0 >= Ne) return;
    const int n0 = blockIdx.x * 64;
    const int so = segoff[e];
    const int tid = threadIdx.x;
    const int lane = tid & 31, wid = tid >> 5;
    const int lr = lane & 15, hf = lane >> 4;
    const int srow = tid >> 1, sseg = (tid & 1) * 16;
    const int soff = srow * 32 + sseg;

    if (tid < 64) {
        int gm = m0 + tid;
        if (gm < Ne) {
            int t = perm[so + gm];
            toks[tid] = t;
            ceL[tid] = combine[(size_t)t * E_EXP + e];
        } else {
            toks[tid] = perm[so];
            ceL[tid] = 0.0f;
        }
    }
    __syncthreads();

    const unsigned short* arow_src = hb + (size_t)(so + m0 + srow) * I_DIM + sseg;
    const float* brow_src = w2 + ((size_t)e * H_DIM + (n0 + srow)) * I_DIM + sseg;

    v8f zero = {};
    v8f acc[4];
#pragma unroll
    for (int i = 0; i < 4; ++i) acc[i] = zero;

    // prologue
    stage16h(arow_src, &As[0][soff]);
    {
        float4 a, b, c, d;
        loadB16f(brow_src, a, b, c, d); storeB16(&Bs[0][soff], a, b, c, d);
    }
    async_fence();
    __syncthreads();

    int buf = 0;
    for (int k0 = 0; k0 < I_DIM; k0 += 32, buf ^= 1) {
        const int kn = k0 + 32;
        const bool hasnext = kn < I_DIM;
        float4 b0, b1, b2, b3;
        if (hasnext) {
            stage16h(arow_src + kn, &As[buf ^ 1][soff]);
            loadB16f(brow_src + kn, b0, b1, b2, b3);
            if (kn + 32 < I_DIM) __builtin_prefetch(brow_src + kn + 32, 0, 3);
        }

        FragBF af;
        const int arow = wid * 16 + lr;
#pragma unroll
        for (int vr = 0; vr < 8; ++vr)
            af.u[vr] = *(const unsigned int*)&As[buf][arow * 32 + kmap(vr, hf)];
#pragma unroll
        for (int nb = 0; nb < 4; ++nb) {
            FragBF bf;
            const int brow = nb * 16 + lr;
#pragma unroll
            for (int vr = 0; vr < 8; ++vr)
                bf.u[vr] = *(const unsigned int*)&Bs[buf][brow * 32 + kmap(vr, hf)];
            acc[nb] = __builtin_amdgcn_wmma_f32_16x16x32_bf16(
                false, af.v, false, bf.v, (short)0, acc[nb], false, false);
        }

        if (hasnext) storeB16(&Bs[buf ^ 1][soff], b0, b1, b2, b3);
        async_fence();
        __syncthreads();
    }

    // scaled scatter-add into y (each token hit by exactly its 2 experts)
#pragma unroll
    for (int nb = 0; nb < 4; ++nb) {
#pragma unroll
        for (int r = 0; r < 8; ++r) {
            int row = wid * 16 + r + 8 * hf;
            int gm = m0 + row;
            if (gm < Ne) {
                atomicAdd(&y[(size_t)toks[row] * H_DIM + (n0 + nb * 16 + lr)],
                          ceL[row] * acc[nb][r]);
            }
        }
    }
}

// ---------- launch ----------
static inline size_t alignUp(size_t v, size_t a) { return (v + a - 1) & ~(a - 1); }

extern "C" void kernel_launch(void* const* d_in, const int* in_sizes, int n_in,
                              void* d_out, int out_size, void* d_ws, size_t ws_size,
                              hipStream_t stream) {
    const float* x    = (const float*)d_in[0];   // [T, H]
    const float* wqkv = (const float*)d_in[1];   // [H, 48]
    const float* w1   = (const float*)d_in[2];   // [E, 2I, H]
    const float* w2   = (const float*)d_in[3];   // [E, H, I]
    float* y = (float*)d_out;                    // [T, H]

    char* ws = (char*)d_ws;
    size_t off = 0;
    float* mix     = (float*)(ws + off); off = alignUp(off + (size_t)T_TOK * 48 * 4, 256);
    float* combine = (float*)(ws + off); off = alignUp(off + (size_t)T_TOK * E_EXP * 4, 256);
    int*   cnt     = (int*)(ws + off);   off = alignUp(off + E_EXP * 4, 256);
    int*   segoff  = (int*)(ws + off);   off = alignUp(off + E_EXP * 4, 256);
    int*   cur     = (int*)(ws + off);   off = alignUp(off + E_EXP * 4, 256);
    int*   ids     = (int*)(ws + off);   off = alignUp(off + (size_t)2 * T_TOK * 4, 256);
    int*   perm    = (int*)(ws + off);   off = alignUp(off + (size_t)2 * T_TOK * 4, 256);
    unsigned short* xb = (unsigned short*)(ws + off);
    off = alignUp(off + (size_t)T_TOK * H_DIM * 2, 256);
    unsigned short* hb = (unsigned short*)(ws + off);
    off = alignUp(off + (size_t)(2 * T_TOK + 64) * I_DIM * 2, 256);
    (void)ws_size; (void)in_sizes; (void)n_in; (void)out_size;

    const int nxh = T_TOK * H_DIM;
    init_kernel<<<(nxh + 255) / 256, 256, 0, stream>>>(y, cnt, nxh);
    cvt_kernel<<<(nxh + 255) / 256, 256, 0, stream>>>(x, xb, nxh);
    mix_kernel<<<(T_TOK * 48 + 255) / 256, 256, 0, stream>>>(x, wqkv, mix);
    router_kernel<<<(T_TOK + 255) / 256, 256, 0, stream>>>(mix, combine, ids, cnt);
    scan_kernel<<<1, 32, 0, stream>>>(cnt, segoff, cur);
    fill_kernel<<<(T_TOK + 255) / 256, 256, 0, stream>>>(ids, cur, perm);
    gemm1_kernel<<<dim3(I_DIM / 64, T_TOK / 64, E_EXP), 128, 0, stream>>>(
        xb, w1, perm, segoff, cnt, hb);
    gemm2_kernel<<<dim3(H_DIM / 64, T_TOK / 64, E_EXP), 128, 0, stream>>>(
        hb, w2, perm, segoff, cnt, combine, y);
}
